// GraphMesh2ConvClassifier_8117488190083
// MI455X (gfx1250) — matrix operand order repaired and checked by reading, not compiled, not run on hardware
//
#include <hip/hip_runtime.h>
#include <hip/hip_bf16.h>

typedef __attribute__((ext_vector_type(16))) _Float16 v16h;
typedef __attribute__((ext_vector_type(8)))  float    v8f;

#define N_NODES   50000
#define N_EDGES   800000
#define N_GRAPHS  16
#define NEG_SLOPE 0.01f

__device__ __forceinline__ float clip_rsqrt(float d) {
    return rsqrtf(fmaxf(d, 1.0f));
}

// ---------------- degree computation ----------------
__global__ void k_degrees(const int* __restrict__ src, const int* __restrict__ dst,
                          float* __restrict__ deg_out, float* __restrict__ deg_in) {
    int t = blockIdx.x * blockDim.x + threadIdx.x;
    if (t < N_EDGES) {
        atomicAdd(&deg_out[src[t]], 1.0f);
        atomicAdd(&deg_in[dst[t]], 1.0f);
    }
}

// ---------------- h0 = features * rsqrt(deg_out) ----------------
__global__ void k_prescale(const float* __restrict__ x, const float* __restrict__ deg_out,
                           float* __restrict__ h0) {
    int t = blockIdx.x * blockDim.x + threadIdx.x;   // N_NODES*64 exact
    int n = t >> 6;
    h0[t] = x[t] * clip_rsqrt(deg_out[n]);
}

// ---------------- scatter-add over edges: agg[dst] += h[src] ----------------
// one warp per edge; each lane handles C/32 channels (L2-resident atomics)
template <int C>
__global__ void k_scatter(const int* __restrict__ src, const int* __restrict__ dst,
                          const float* __restrict__ h, float* __restrict__ agg) {
    const int lane = threadIdx.x & 31;
    const int warp = threadIdx.x >> 5;
    const int e = blockIdx.x * (blockDim.x >> 5) + warp;
    if (e >= N_EDGES) return;
    const int s = src[e], d = dst[e];
    const float* hs = h + (size_t)s * C;
    float* ad = agg + (size_t)d * C;
#pragma unroll
    for (int i = 0; i < C / 32; ++i) {
        int c = lane + i * 32;
        atomicAdd(&ad[c], hs[c]);
    }
}

// ---------------- pre-pack weights into WMMA B-fragment order (f16) ----------------
// B fragment layout (ISA 7.12.2): lane -> col = tn*16 + (lane&15), hi = lane>>4;
// element e -> K = kc*32 + (e<8 ? hi*8 + e : 16 + hi*8 + (e-8)).
// Packed layout: P[(((tn*nchunks)+kc)*32 + lane)*16 + e]  -> one contiguous 32B per lane.
__global__ void k_pack_b(const float* __restrict__ B, _Float16* __restrict__ P,
                         int ntiles, int nchunks, int ldb, int transposed) {
    int t = blockIdx.x * blockDim.x + threadIdx.x;
    int total = ntiles * nchunks * 32 * 16;
    if (t >= total) return;
    int e    = t & 15;
    int lane = (t >> 4) & 31;
    int kc   = (t >> 9) % nchunks;
    int tn   = t / (nchunks * 512);
    int hi = lane >> 4, l16 = lane & 15;
    int col = tn * 16 + l16;
    int k = kc * 32 + ((e < 8) ? (hi * 8 + e) : (16 + hi * 8 + (e - 8)));
    float v = transposed ? B[(size_t)col * ldb + k] : B[(size_t)k * ldb + col];
    P[t] = (_Float16)v;
}

// ---------------- A-fragment loader (ISA 7.12.2 layout) ----------------
// A 16x32 f16: lane holds row = lane&15; VGPR j (pair 2j,2j+1):
// basek = (j<4 ? 2j : 16+2(j-4)) + 8*(lane>>4); contiguous 32B chunks -> vectorized
__device__ __forceinline__ v16h load_a_f32(const float* __restrict__ A, int lda,
                                           int row, int hi, int kb, float scale) {
    v16h a;
    const float* Ar = A + (size_t)row * lda + kb + hi * 8;
#pragma unroll
    for (int j = 0; j < 8; ++j) {
        int basek = (j < 4) ? (2 * j) : (16 + 2 * (j - 4));
        a[2 * j]     = (_Float16)(Ar[basek] * scale);
        a[2 * j + 1] = (_Float16)(Ar[basek + 1] * scale);
    }
    return a;
}

__device__ __forceinline__ v16h load_b_packed(const _Float16* __restrict__ P,
                                              int nchunks, int tn, int kc, int lane) {
    return *(const v16h*)(P + (size_t)(((tn * nchunks) + kc) * 32 + lane) * 16);
}

// ---------------- GEMM1: h1s = leaky((agg0 * rs_in) @ W1) * rs_out ----------------
// agg0 [N,64] @ W1 [64,128] -> h1s [N,128]; one wave per 16x16 tile
__global__ void k_gcn_gemm1(const float* __restrict__ agg0, const _Float16* __restrict__ W1p,
                            const float* __restrict__ deg_in, const float* __restrict__ deg_out,
                            float* __restrict__ h1s) {
    const int lane = threadIdx.x;
    const int hi = lane >> 4, l16 = lane & 15;
    const int tm = blockIdx.x;                               // 3125 row tiles (exact)
    const int tn = blockIdx.y * blockDim.y + threadIdx.y;    // 8 col tiles
    const int arow = tm * 16 + l16;
    const float rs_in = clip_rsqrt(deg_in[arow]);
    const int ncol = tn * 16 + l16;
    v8f acc = {};
#pragma unroll
    for (int kc = 0; kc < 2; ++kc) {
        v16h a = load_a_f32(agg0, 64, arow, hi, kc * 32, rs_in);
        v16h b = load_b_packed(W1p, 2, tn, kc, lane);
        acc = __builtin_amdgcn_wmma_f32_16x16x32_f16(false, a, false, b,
                                                     (short)0, acc, false, false);
    }
#pragma unroll
    for (int v = 0; v < 8; ++v) {
        int m = tm * 16 + hi * 8 + v;                        // C layout: M = v + 8*hi
        float x = acc[v];
        x = (x >= 0.0f) ? x : NEG_SLOPE * x;                 // LeakyReLU
        h1s[(size_t)m * 128 + ncol] = x * clip_rsqrt(deg_out[m]);  // fold layer-2 pre-scale
    }
}

// ---------------- GEMM2: h2 = leaky((agg1 * rs_in) @ W2) ----------------
// agg1 [N,128] @ W2 [128,64] -> h2 [N,64]
__global__ void k_gcn_gemm2(const float* __restrict__ agg1, const _Float16* __restrict__ W2p,
                            const float* __restrict__ deg_in, float* __restrict__ h2) {
    const int lane = threadIdx.x;
    const int hi = lane >> 4, l16 = lane & 15;
    const int tm = blockIdx.x;
    const int tn = threadIdx.y;                              // 4 col tiles
    const int arow = tm * 16 + l16;
    const float rs_in = clip_rsqrt(deg_in[arow]);
    const int ncol = tn * 16 + l16;
    v8f acc = {};
#pragma unroll
    for (int kc = 0; kc < 4; ++kc) {
        v16h a = load_a_f32(agg1, 128, arow, hi, kc * 32, rs_in);
        v16h b = load_b_packed(W2p, 4, tn, kc, lane);
        acc = __builtin_amdgcn_wmma_f32_16x16x32_f16(false, a, false, b,
                                                     (short)0, acc, false, false);
    }
#pragma unroll
    for (int v = 0; v < 8; ++v) {
        int m = tm * 16 + hi * 8 + v;
        float x = acc[v];
        h2[(size_t)m * 64 + ncol] = (x >= 0.0f) ? x : NEG_SLOPE * x;
    }
}

// ---------------- Linear: h3 = h2 @ linW.T + linb ----------------
// h2 [N,64] @ linW.T [64,32] -> h3 [N,32]
__global__ void k_lin(const float* __restrict__ h2, const _Float16* __restrict__ linWp,
                      const float* __restrict__ linb, float* __restrict__ h3) {
    const int lane = threadIdx.x;
    const int hi = lane >> 4, l16 = lane & 15;
    const int tm = blockIdx.x;
    const int tn = threadIdx.y;                              // 2 col tiles
    const int arow = tm * 16 + l16;
    const int ncol = tn * 16 + l16;
    v8f acc = {};
#pragma unroll
    for (int kc = 0; kc < 2; ++kc) {
        v16h a = load_a_f32(h2, 64, arow, hi, kc * 32, 1.0f);
        v16h b = load_b_packed(linWp, 2, tn, kc, lane);
        acc = __builtin_amdgcn_wmma_f32_16x16x32_f16(false, a, false, b,
                                                     (short)0, acc, false, false);
    }
    const float bias = linb[ncol];
#pragma unroll
    for (int v = 0; v < 8; ++v) {
        int m = tm * 16 + hi * 8 + v;
        h3[(size_t)m * 32 + ncol] = acc[v] + bias;
    }
}

// ---------------- per-graph pooling (segment sums + counts) ----------------
__global__ void k_pool(const float* __restrict__ h3, const int* __restrict__ gid,
                       float* __restrict__ pooled, float* __restrict__ counts) {
    int t = blockIdx.x * blockDim.x + threadIdx.x;   // N_NODES*32 exact
    int n = t >> 5, c = t & 31;
    int g = gid[n];
    atomicAdd(&pooled[g * 32 + c], h3[t]);
    if (c == 0) atomicAdd(&counts[g], 1.0f);
}

// ---------------- classifier: out = (pooled/counts) @ clsW.T ----------------
__global__ void k_cls(const float* __restrict__ pooled, const float* __restrict__ counts,
                      const float* __restrict__ clsW, float* __restrict__ out) {
    int t = threadIdx.x;
    if (t < N_GRAPHS * 10) {
        int b = t / 10, k = t % 10;
        float inv = 1.0f / fmaxf(counts[b], 1.0f);
        float s = 0.0f;
#pragma unroll
        for (int c = 0; c < 32; ++c) s += pooled[b * 32 + c] * clsW[k * 32 + c];
        out[t] = s * inv;
    }
}

extern "C" void kernel_launch(void* const* d_in, const int* in_sizes, int n_in,
                              void* d_out, int out_size, void* d_ws, size_t ws_size,
                              hipStream_t stream) {
    const float* features = (const float*)d_in[0];
    const float* W1   = (const float*)d_in[1];
    const float* W2   = (const float*)d_in[2];
    const float* linW = (const float*)d_in[3];
    const float* linb = (const float*)d_in[4];
    const float* clsW = (const float*)d_in[5];
    const int*   src  = (const int*)d_in[6];
    const int*   dst  = (const int*)d_in[7];
    const int*   gid  = (const int*)d_in[8];
    float* out = (float*)d_out;

    // workspace layout (float offsets); buffers aliased once producers retire
    float* ws      = (float*)d_ws;
    float* deg_out = ws;                   // 50000
    float* deg_in  = ws + 50000;           // 50000
    float* pooled  = ws + 100000;          // 16*32
    float* counts  = ws + 100512;          // 16
    _Float16* W1p   = (_Float16*)(ws + 100544);   //  8 tiles * 2 chunks * 512 = 8192 halfs
    _Float16* W2p   = W1p + 8192;                 //  4 tiles * 4 chunks * 512 = 8192 halfs
    _Float16* linWp = W2p + 8192;                 //  2 tiles * 2 chunks * 512 = 2048 halfs
    float* reg1 = ws + 110080;             // 6.4M floats
    float* reg2 = reg1 + 6400000;          // 6.4M floats
    float* h0   = reg1;                    // [N,64]
    float* agg0 = reg1 + 3200000;          // [N,64]
    float* h1s  = reg2;                    // [N,128]
    float* agg1 = reg1;                    // [N,128] (reuses h0+agg0, both dead)
    float* h2   = reg2;                    // [N,64]  (reuses h1s, dead)
    float* h3   = reg2 + 3200000;          // [N,32]

    hipMemsetAsync(deg_out, 0, 2 * 50000 * sizeof(float), stream);
    hipMemsetAsync(pooled, 0, (512 + 16) * sizeof(float), stream);

    // pack weights into f16 WMMA fragment order (tiny, done every call: deterministic)
    k_pack_b<<<32, 256, 0, stream>>>(W1, W1p, 8, 2, 128, 0);
    k_pack_b<<<32, 256, 0, stream>>>(W2, W2p, 4, 4, 64, 0);
    k_pack_b<<<8, 256, 0, stream>>>(linW, linWp, 2, 2, 64, 1);

    k_degrees<<<(N_EDGES + 255) / 256, 256, 0, stream>>>(src, dst, deg_out, deg_in);
    k_prescale<<<(N_NODES * 64) / 256, 256, 0, stream>>>(features, deg_out, h0);

    hipMemsetAsync(agg0, 0, (size_t)N_NODES * 64 * sizeof(float), stream);
    k_scatter<64><<<N_EDGES / 8, 256, 0, stream>>>(src, dst, h0, agg0);

    k_gcn_gemm1<<<dim3(N_NODES / 16, 2), dim3(32, 4), 0, stream>>>(agg0, W1p, deg_in, deg_out, h1s);

    hipMemsetAsync(agg1, 0, (size_t)N_NODES * 128 * sizeof(float), stream);
    k_scatter<128><<<N_EDGES / 8, 256, 0, stream>>>(src, dst, h1s, agg1);

    k_gcn_gemm2<<<dim3(N_NODES / 16, 1), dim3(32, 4), 0, stream>>>(agg1, W2p, deg_in, h2);
    k_lin<<<dim3(N_NODES / 16, 1), dim3(32, 2), 0, stream>>>(h2, linWp, linb, h3);

    k_pool<<<(N_NODES * 32) / 256, 256, 0, stream>>>(h3, gid, pooled, counts);
    k_cls<<<1, 256, 0, stream>>>(pooled, counts, clsW, out);
}
